// GroupedMLP_PlanA_55216099557792
// MI455X (gfx1250) — compile-verified, matrix operand order checked by reading, not run on hardware
//
#include <hip/hip_runtime.h>
#include <hip/hip_bf16.h>

// Grouped SwiGLU MLP for MI455X (gfx1250), bf16 WMMA path.
// G=8, D_IN=1024, D_HID=4096, D_OUT=1024, TOKENS=16384 (B=2048 per expert).

#define G_EXPERTS 8
#define D_IN      1024
#define D_HID     4096
#define D_OUT     1024
#define TOKENS    16384
#define BATCH     (TOKENS / G_EXPERTS)   // 2048

#define BM 128
#define BN 128
#define BK 32
#define LDA 36   // ushort stride, BK + 4 pad (keeps 4/8-byte alignment, skews banks)
#define LDB 36

typedef __attribute__((ext_vector_type(16))) __bf16 v16bf;
typedef __attribute__((ext_vector_type(8)))  float  v8f;
typedef __attribute__((ext_vector_type(8)))  int    v8i;

union FragU { v8i i; v16bf bf; };

// f32 -> bf16 round-to-nearest-even (bit pattern as ushort)
__device__ __forceinline__ unsigned short f2bf(float f) {
  unsigned int u = __builtin_bit_cast(unsigned int, f);
  u += 0x7FFFu + ((u >> 16) & 1u);
  return (unsigned short)(u >> 16);
}

// silu(x) = x * sigmoid(x), sigmoid via hardware v_rcp_f32 (avoids IEEE div chain)
__device__ __forceinline__ float silu_fast(float x) {
  return x * __builtin_amdgcn_rcpf(1.0f + __expf(-x));
}

// K index for 16-bit WMMA fragment VGPR i (pair base), given lane half (lane>>4).
// ISA 05_wmma: 16-bit A 16x32 -> VGPR0..3: K=0..7 (half 0) / 8..15 (half 1),
// VGPR4..7: K=16..23 / 24..31. B (32x16) assumed symmetric: lane = column N.
__device__ __forceinline__ int kmap(int i, int half) {
  return (i < 4) ? (2 * i + 8 * half) : (16 + 2 * (i - 4) + 8 * half);
}

// Load one 16x32 (or 32x16 transposed-in-LDS) bf16 fragment: 8 aligned 32-bit LDS reads.
__device__ __forceinline__ v16bf load_frag(const unsigned short* base, int row, int ld, int half) {
  FragU f;
#pragma unroll
  for (int i = 0; i < 8; ++i) {
    int k = kmap(i, half);
    f.i[i] = *(const int*)(base + row * ld + k);   // k even, row*ld*2 % 4 == 0
  }
  return f.bf;
}

// ---------------------------------------------------------------------------
// Kernel 1: hidden[t, n] = silu(x@gate) * (x@up), bf16 output to workspace.
// Grid: (D_HID/BN, BATCH/BM, G). Block: 256 threads = 8 waves (wave32).
// ---------------------------------------------------------------------------
__global__ __launch_bounds__(256)
void gateup_swiglu_kernel(const float* __restrict__ x,
                          const float* __restrict__ gw,
                          const float* __restrict__ uw,
                          unsigned short* __restrict__ hid) {
  __shared__ unsigned short sA [BM * LDA];
  __shared__ unsigned short sBg[BN * LDB];
  __shared__ unsigned short sBu[BN * LDB];

  const int g   = blockIdx.z;
  const int m0  = blockIdx.y * BM;   // token index within expert (b)
  const int n0  = blockIdx.x * BN;   // hidden column

  const int tid  = threadIdx.x;
  const int lane = tid & 31;
  const int wid  = tid >> 5;
  const int wm   = wid & 3;          // 4 waves over M: 32 rows each
  const int wn   = wid >> 2;         // 2 waves over N: 64 cols each
  const int half = lane >> 4;
  const int l16  = lane & 15;

  const float* gwg = gw + (size_t)g * D_IN * D_HID;
  const float* uwg = uw + (size_t)g * D_IN * D_HID;

  v8f accg[2][4] = {};
  v8f accu[2][4] = {};

  for (int k0 = 0; k0 < D_IN; k0 += BK) {
    __syncthreads();
    // A tile: 128x32 f32 -> bf16 LDS, row-major [m][k]. 4 float4 per thread.
#pragma unroll
    for (int i = 0; i < 4; ++i) {
      int li = tid + i * 256;        // 0..1023 float4 slots
      int r  = li >> 3;
      int c4 = (li & 7) * 4;
      const float4 v = *(const float4*)(x + ((size_t)(m0 + r) * G_EXPERTS + g) * D_IN + k0 + c4);
      unsigned int p0 = ((unsigned int)f2bf(v.y) << 16) | f2bf(v.x);
      unsigned int p1 = ((unsigned int)f2bf(v.w) << 16) | f2bf(v.z);
      *(unsigned int*)(&sA[r * LDA + c4])     = p0;
      *(unsigned int*)(&sA[r * LDA + c4 + 2]) = p1;
    }
    // B tiles: 32x128 f32 each, transposed into LDS [n][k].
#pragma unroll
    for (int i = 0; i < 4; ++i) {
      int li = tid + i * 256;
      int kk = li >> 5;              // 0..31
      int nn = (li & 31) * 4;        // 0..124
      const float4 vg = *(const float4*)(gwg + (size_t)(k0 + kk) * D_HID + n0 + nn);
      const float4 vu = *(const float4*)(uwg + (size_t)(k0 + kk) * D_HID + n0 + nn);
      sBg[(nn + 0) * LDB + kk] = f2bf(vg.x);
      sBg[(nn + 1) * LDB + kk] = f2bf(vg.y);
      sBg[(nn + 2) * LDB + kk] = f2bf(vg.z);
      sBg[(nn + 3) * LDB + kk] = f2bf(vg.w);
      sBu[(nn + 0) * LDB + kk] = f2bf(vu.x);
      sBu[(nn + 1) * LDB + kk] = f2bf(vu.y);
      sBu[(nn + 2) * LDB + kk] = f2bf(vu.z);
      sBu[(nn + 3) * LDB + kk] = f2bf(vu.w);
    }
    __syncthreads();

    v16bf afr[2];
    afr[0] = load_frag(sA, wm * 32 +  0 + l16, LDA, half);
    afr[1] = load_frag(sA, wm * 32 + 16 + l16, LDA, half);
#pragma unroll
    for (int nf = 0; nf < 4; ++nf) {
      int nrow = wn * 64 + nf * 16 + l16;
      v16bf bg = load_frag(sBg, nrow, LDB, half);
      v16bf bu = load_frag(sBu, nrow, LDB, half);
#pragma unroll
      for (int mf = 0; mf < 2; ++mf) {
        accg[mf][nf] = __builtin_amdgcn_wmma_f32_16x16x32_bf16(
            false, afr[mf], false, bg, (short)0, accg[mf][nf], false, false);
        accu[mf][nf] = __builtin_amdgcn_wmma_f32_16x16x32_bf16(
            false, afr[mf], false, bu, (short)0, accu[mf][nf], false, false);
      }
    }
  }

  // Epilogue: silu(g) * u -> bf16 hidden at row (b*G + g).
  // C/D layout: VGPR r -> M = r + 8*(lane>>4), N = lane&15.
#pragma unroll
  for (int mf = 0; mf < 2; ++mf) {
#pragma unroll
    for (int nf = 0; nf < 4; ++nf) {
#pragma unroll
      for (int r = 0; r < 8; ++r) {
        int m = m0 + wm * 32 + mf * 16 + r + 8 * half;
        int n = n0 + wn * 64 + nf * 16 + l16;
        float h = silu_fast(accg[mf][nf][r]) * accu[mf][nf][r];
        hid[((size_t)m * G_EXPERTS + g) * D_HID + n] = f2bf(h);
      }
    }
  }
}

// ---------------------------------------------------------------------------
// Kernel 2: out[t, o] = hidden @ down (bf16 x bf16 -> f32).
// Grid: (D_OUT/BN, BATCH/BM, G). Block: 256 threads.
// ---------------------------------------------------------------------------
__global__ __launch_bounds__(256)
void down_kernel(const unsigned short* __restrict__ hid,
                 const float* __restrict__ dw,
                 float* __restrict__ out) {
  __shared__ unsigned short sA[BM * LDA];
  __shared__ unsigned short sB[BN * LDB];

  const int g  = blockIdx.z;
  const int m0 = blockIdx.y * BM;
  const int n0 = blockIdx.x * BN;

  const int tid  = threadIdx.x;
  const int lane = tid & 31;
  const int wid  = tid >> 5;
  const int wm   = wid & 3;
  const int wn   = wid >> 2;
  const int half = lane >> 4;
  const int l16  = lane & 15;

  const float* dwg = dw + (size_t)g * D_HID * D_OUT;

  v8f acc[2][4] = {};

  for (int k0 = 0; k0 < D_HID; k0 += BK) {
    __syncthreads();
    // A tile: 128x32 bf16 (already bf16 in ws). 4 x uint2 (4 ushorts) per thread.
#pragma unroll
    for (int i = 0; i < 4; ++i) {
      int li = tid + i * 256;
      int r  = li >> 3;
      int c4 = (li & 7) * 4;
      const uint2 v = *(const uint2*)(hid + ((size_t)(m0 + r) * G_EXPERTS + g) * D_HID + k0 + c4);
      *(uint2*)(&sA[r * LDA + c4]) = v;   // 8-byte aligned: 72*r + 2*c4 ≡ 0 (mod 8)
    }
    // B tile: 32x128 f32 down-weight, transposed into LDS [n][k].
#pragma unroll
    for (int i = 0; i < 4; ++i) {
      int li = tid + i * 256;
      int kk = li >> 5;
      int nn = (li & 31) * 4;
      const float4 v = *(const float4*)(dwg + (size_t)(k0 + kk) * D_OUT + n0 + nn);
      sB[(nn + 0) * LDB + kk] = f2bf(v.x);
      sB[(nn + 1) * LDB + kk] = f2bf(v.y);
      sB[(nn + 2) * LDB + kk] = f2bf(v.z);
      sB[(nn + 3) * LDB + kk] = f2bf(v.w);
    }
    __syncthreads();

    v16bf afr[2];
    afr[0] = load_frag(sA, wm * 32 +  0 + l16, LDA, half);
    afr[1] = load_frag(sA, wm * 32 + 16 + l16, LDA, half);
#pragma unroll
    for (int nf = 0; nf < 4; ++nf) {
      v16bf bf = load_frag(sB, wn * 64 + nf * 16 + l16, LDB, half);
#pragma unroll
      for (int mf = 0; mf < 2; ++mf) {
        acc[mf][nf] = __builtin_amdgcn_wmma_f32_16x16x32_bf16(
            false, afr[mf], false, bf, (short)0, acc[mf][nf], false, false);
      }
    }
  }

#pragma unroll
  for (int mf = 0; mf < 2; ++mf) {
#pragma unroll
    for (int nf = 0; nf < 4; ++nf) {
#pragma unroll
      for (int r = 0; r < 8; ++r) {
        int m = m0 + wm * 32 + mf * 16 + r + 8 * half;
        int n = n0 + wn * 64 + nf * 16 + l16;
        out[((size_t)m * G_EXPERTS + g) * D_OUT + n] = acc[mf][nf][r];
      }
    }
  }
}

extern "C" void kernel_launch(void* const* d_in, const int* in_sizes, int n_in,
                              void* d_out, int out_size, void* d_ws, size_t ws_size,
                              hipStream_t stream) {
  const float* x  = (const float*)d_in[0];
  const float* gw = (const float*)d_in[1];
  const float* uw = (const float*)d_in[2];
  const float* dw = (const float*)d_in[3];
  unsigned short* hid = (unsigned short*)d_ws;   // TOKENS*D_HID bf16 = 128 MB

  dim3 blk(256, 1, 1);
  dim3 g1(D_HID / BN, BATCH / BM, G_EXPERTS);    // 32 x 16 x 8
  gateup_swiglu_kernel<<<g1, blk, 0, stream>>>(x, gw, uw, hid);

  dim3 g2(D_OUT / BN, BATCH / BM, G_EXPERTS);    // 8 x 16 x 8
  down_kernel<<<g2, blk, 0, stream>>>(hid, dw, (float*)d_out);
}
